// MultiScaleAttention_36240934043808
// MI455X (gfx1250) — matrix operand order repaired and checked by reading, not compiled, not run on hardware
//
#include <hip/hip_runtime.h>
#include <hip/hip_bf16.h>

// ---------------- problem constants ----------------
#define BB      4
#define TT      8
#define HH      14
#define WW      14
#define NTOK    1569            // T*H*W + 1
#define LSP     1568            // spatial tokens
#define CIN     768
#define COUT    768
#define HEADS   12
#define HD      64
#define QKV3    2304
#define MTOT    (BB*NTOK)       // 6276 rows
#define MTILES  ((MTOT+15)/16)  // 393
#define SCALE_F 0.125f          // 64^-0.5
#define NEGBIG  (-1.0e30f)

typedef __attribute__((ext_vector_type(16))) _Float16 v16h;
typedef __attribute__((ext_vector_type(16))) __bf16   v16bf;
typedef __attribute__((ext_vector_type(8)))  __bf16   v8bf;
typedef __attribute__((ext_vector_type(8)))  float    v8f;

// ---------------- WMMA fragment loaders (wave32; contiguous k = 16h + s, not the ISA 7.12.2 table order) ----
// A (16x32, 16-bit), k order used here: lane 0-15 row=lane, K elems {0..15}; lane 16-31 row=lane-16, K {16..31}
__device__ __forceinline__ v16bf load_a_bf(const __bf16* base, int ld, int row0,
                                           int k0, int rowMax) {
  int lane = threadIdx.x & 31;
  int r = row0 + (lane & 15);
  if (r >= rowMax) r = rowMax - 1;
  int kh = (lane >> 4) * 16;
  const __bf16* p = base + (size_t)r * ld + k0 + kh;
  v8bf c0 = *(const v8bf*)(p);
  v8bf c1 = *(const v8bf*)(p + 8);
  v16bf a;
#pragma unroll
  for (int i = 0; i < 8; i++) { a[i] = c0[i]; a[8 + i] = c1[i]; }
  return a;
}

// B (32x16, 16-bit): lane holds column n=lane&15; K = 16*(lane>>4) .. +15 contiguous.
// Source must be column-contiguous: row-major [col][k] with stride ld.
__device__ __forceinline__ v16bf load_b_bf(const __bf16* base, int ld, int col0,
                                           int k0, int colMax) {
  int lane = threadIdx.x & 31;
  int c = col0 + (lane & 15);
  if (c >= colMax) c = colMax - 1;
  int kc = (lane >> 4) * 16;
  const __bf16* p = base + (size_t)c * ld + k0 + kc;
  v8bf c0 = *(const v8bf*)(p);
  v8bf c1 = *(const v8bf*)(p + 8);
  v16bf b;
#pragma unroll
  for (int i = 0; i < 8; i++) { b[i] = c0[i]; b[8 + i] = c1[i]; }
  return b;
}

#define WMMA_BF(A, B, C) \
  __builtin_amdgcn_wmma_f32_16x16x32_bf16(false, (A), false, (B), (short)0, (C), false, false)
#define WMMA_F16(A, B, C) \
  __builtin_amdgcn_wmma_f32_16x16x32_f16(false, (A), false, (B), (short)0, (C), false, false)

// Async global->LDS stage of one contiguous 4KB tile: 8 issues x 32 lanes x 16B.
#define ASYNC_TILE_4K(gbase, ldsoff)                                            \
  {                                                                             \
    _Pragma("unroll")                                                           \
    for (int i_ = 0; i_ < 8; i_++) {                                            \
      unsigned lo_ = (unsigned)(ldsoff) + (unsigned)(lane * 16 + i_ * 512);     \
      const char* ga_ = (const char*)(gbase) + lane * 16 + i_ * 512;            \
      asm volatile("global_load_async_to_lds_b128 %0, %1, off"                  \
                   :: "v"(lo_), "v"(ga_) : "memory");                           \
    }                                                                           \
  }

// ---------------- prep kernels ----------------
__global__ void cvt_bf16_kernel(const float* __restrict__ in, __bf16* __restrict__ out, int n) {
  int i = blockIdx.x * blockDim.x + threadIdx.x;
  if (i < n) out[i] = (__bf16)in[i];
}

// in: rows x cols (row-major fp32)  ->  out: cols x rows (row-major bf16)
__global__ void transpose_bf16_kernel(const float* __restrict__ in, __bf16* __restrict__ out,
                                      int rows, int cols) {
  int i = blockIdx.x * blockDim.x + threadIdx.x;
  if (i < rows * cols) {
    int cc = i / rows, rr = i % rows;
    out[i] = (__bf16)in[(size_t)rr * cols + cc];
  }
}

// ---------------- QKV GEMM: (6276x768)@(768x2304)+b, scatter into [B][HEADS][N][HD] ----
__global__ __launch_bounds__(32) void qkv_gemm_kernel(
    const __bf16* __restrict__ xb,   // 6276x768 bf16
    const __bf16* __restrict__ wt,   // 2304x768 bf16 (pre-transposed)
    const float*  __restrict__ bias, // 2304
    float* __restrict__ qraw, float* __restrict__ kraw, float* __restrict__ vraw) {
  int idx = blockIdx.x;
  int mt = idx / 36;          // 393 M tiles
  int ng = idx % 36;          // 36 groups of 64 columns
  int lane = threadIdx.x;
  int hl = lane >> 4;
  int col0 = ng * 64;
  v8f acc[4] = {v8f{}, v8f{}, v8f{}, v8f{}};
  for (int k0 = 0; k0 < 768; k0 += 32) {
    __builtin_prefetch(xb + (size_t)(mt * 16 + (lane & 15)) * 768 + k0 + 64, 0, 1);
    v16bf a = load_a_bf(xb, 768, mt * 16, k0, MTOT);
#pragma unroll
    for (int j = 0; j < 4; j++) {
      v16bf b = load_b_bf(wt, 768, col0 + j * 16, k0, QKV3);
      acc[j] = WMMA_BF(a, b, acc[j]);
    }
  }
#pragma unroll
  for (int j = 0; j < 4; j++) {
    int ncol = col0 + j * 16 + (lane & 15);
    int which = ncol / 768;
    int head = (ncol % 768) / 64;
    int hd = ncol & 63;
    float bi = bias[ncol];
    float* dst = (which == 0) ? qraw : ((which == 1) ? kraw : vraw);
#pragma unroll
    for (int r = 0; r < 8; r++) {
      int m = mt * 16 + r + 8 * hl;
      if (m < MTOT) {
        int b = m / NTOK, tok = m % NTOK;
        dst[(((size_t)b * HEADS + head) * NTOK + tok) * HD + hd] = acc[j][r] + bi;
      }
    }
  }
}

// ---------------- depthwise 3x3x3 conv + LayerNorm(HD) ----------------
__global__ __launch_bounds__(64) void pool_ln_kernel(
    const float* __restrict__ raw,      // [BH][NTOK][HD]
    const float* __restrict__ cw,       // [HD][3][3][3]
    const float* __restrict__ nw, const float* __restrict__ nb,
    float* __restrict__ poolf,          // fp32 out
    __bf16* __restrict__ pool_bf,       // optional bf16 copy
    _Float16* __restrict__ pool_h) {    // optional f16 copy
  int blk = blockIdx.x;
  int tok = blk % NTOK;
  int bh = blk / NTOK;
  int c = threadIdx.x;
  const float* src = raw + (size_t)bh * NTOK * HD;
  float val;
  if (tok == 0) {
    val = src[c];
  } else {
    int l = tok - 1;
    int t = l / (HH * WW), rem = l % (HH * WW), y = rem / WW, x = rem % WW;
    float s = 0.f;
#pragma unroll
    for (int dz = -1; dz <= 1; dz++) {
      int tt = t + dz; if (tt < 0 || tt >= TT) continue;
#pragma unroll
      for (int dy = -1; dy <= 1; dy++) {
        int yy = y + dy; if (yy < 0 || yy >= HH) continue;
#pragma unroll
        for (int dx = -1; dx <= 1; dx++) {
          int xx = x + dx; if (xx < 0 || xx >= WW) continue;
          int tok2 = 1 + (tt * HH + yy) * WW + xx;
          s += src[(size_t)tok2 * HD + c] * cw[c * 27 + (dz + 1) * 9 + (dy + 1) * 3 + (dx + 1)];
        }
      }
    }
    val = s;
  }
  __shared__ float red[64];
  red[c] = val; __syncthreads();
  for (int s2 = 32; s2 > 0; s2 >>= 1) { if (c < s2) red[c] += red[c + s2]; __syncthreads(); }
  float mean = red[0] * (1.f / 64.f);
  __syncthreads();
  float d = val - mean;
  red[c] = d * d; __syncthreads();
  for (int s2 = 32; s2 > 0; s2 >>= 1) { if (c < s2) red[c] += red[c + s2]; __syncthreads(); }
  float var = red[0] * (1.f / 64.f);
  float o = d * rsqrtf(var + 1e-5f) * nw[c] + nb[c];
  size_t oi = ((size_t)bh * NTOK + tok) * HD + c;
  poolf[oi] = o;
  if (pool_bf) pool_bf[oi] = (__bf16)o;
  if (pool_h)  pool_h[oi]  = (_Float16)o;
}

// ---------------- separable rel-pos dot products ----------------
// relt[bh][l][8], relh[bh][l][14], relw[bh][l][14]
__global__ __launch_bounds__(64) void relbias_kernel(
    const float* __restrict__ qpool,
    const float* __restrict__ rpt, const float* __restrict__ rph, const float* __restrict__ rpw,
    float* __restrict__ relt, float* __restrict__ relh, float* __restrict__ relw) {
  int blk = blockIdx.x;
  int l = blk % LSP;
  int bh = blk / LSP;
  __shared__ float qv[64];
  int tid = threadIdx.x;
  qv[tid] = qpool[((size_t)bh * NTOK + 1 + l) * HD + tid];
  __syncthreads();
  int t = l / (HH * WW), rem = l % (HH * WW), y = rem / WW, x = rem % WW;
  if (tid < 36) {
    const float* tab; int idx; float* dst;
    if (tid < 8)       { tab = rpt; idx = t - tid + (TT - 1);        dst = relt + ((size_t)bh * LSP + l) * 8  + tid; }
    else if (tid < 22) { int j = tid - 8;  tab = rph; idx = y - j + (HH - 1); dst = relh + ((size_t)bh * LSP + l) * 14 + j; }
    else               { int j = tid - 22; tab = rpw; idx = x - j + (WW - 1); dst = relw + ((size_t)bh * LSP + l) * 14 + j; }
    const float* row = tab + (size_t)idx * HD;
    float s = 0.f;
#pragma unroll 8
    for (int c2 = 0; c2 < 64; c2++) s += qv[c2] * row[c2];
    *dst = s;
  }
}

// ---------------- flash attention with separable rel-pos bias ----------------
// K/V 32-key tiles are contiguous 4KB in [key][ch] layout -> staged to LDS with
// double-buffered GLOBAL_LOAD_ASYNC_TO_LDS_B128, synchronized with ASYNCcnt.
#define QTILES 99   // ceil(1569/16)
#define KTILES 50   // ceil(1569/32)
__global__ __launch_bounds__(32) void attn_kernel(
    const __bf16* __restrict__ qbf, const __bf16* __restrict__ kbf,
    const _Float16* __restrict__ vh,
    const float* __restrict__ relt, const float* __restrict__ relh, const float* __restrict__ relw,
    const float* __restrict__ qpool,
    __bf16* __restrict__ outbf) {     // [B][NTOK][768] bf16
  int blk = blockIdx.x;
  int qt = blk % QTILES;
  int bh = blk / QTILES;
  int b = bh / HEADS, head = bh % HEADS;
  int lane = threadIdx.x;
  int hl = lane >> 4, ln = lane & 15;
  const __bf16*   qb = qbf + (size_t)bh * NTOK * HD;
  const __bf16*   kb = kbf + (size_t)bh * NTOK * HD;
  const _Float16* vb = vh  + (size_t)bh * NTOK * HD;
  int q0 = qt * 16;

  __shared__ float relq[16][36];
  __shared__ __align__(16) _Float16 pt[16 * 32];
  __shared__ __align__(16) __bf16   ktile[2][32 * 64];   // 2 x 4KB
  __shared__ __align__(16) _Float16 vtile[2][32 * 64];   // 2 x 4KB

  for (int idx = lane; idx < 16 * 36; idx += 32) {
    int i = idx / 36, j = idx % 36;
    int q = q0 + i;
    float v2 = 0.f;
    if (q >= 1 && q < NTOK) {
      size_t l = (size_t)(q - 1);
      if (j < 8)       v2 = relt[((size_t)bh * LSP + l) * 8 + j];
      else if (j < 22) v2 = relh[((size_t)bh * LSP + l) * 14 + (j - 8)];
      else             v2 = relw[((size_t)bh * LSP + l) * 14 + (j - 22)];
    }
    relq[i][j] = v2;
  }
  __syncthreads();

  v16bf A0 = load_a_bf(qb, HD, q0, 0, NTOK);
  v16bf A1 = load_a_bf(qb, HD, q0, 32, NTOK);

  unsigned k_lds[2], v_lds[2];
  k_lds[0] = (unsigned)(uintptr_t)(&ktile[0][0]);
  k_lds[1] = (unsigned)(uintptr_t)(&ktile[1][0]);
  v_lds[0] = (unsigned)(uintptr_t)(&vtile[0][0]);
  v_lds[1] = (unsigned)(uintptr_t)(&vtile[1][0]);

  // prefetch tile 0 (16 async ops: 8 for K, 8 for V)
  ASYNC_TILE_4K(kb, k_lds[0]);
  ASYNC_TILE_4K(vb, v_lds[0]);

  float mrow[8], lrow[8];
  v8f O[4] = {v8f{}, v8f{}, v8f{}, v8f{}};
#pragma unroll
  for (int r = 0; r < 8; r++) { mrow[r] = NEGBIG; lrow[r] = 0.f; }

  for (int kt = 0; kt < KTILES; kt++) {
    int k0 = kt * 32;
    int cur = kt & 1;
    if (kt + 1 < KTILES) {
      // issue next tile, then wait so only the 16 new ops remain outstanding
      int nk0 = k0 + 32, nb2 = (kt + 1) & 1;
      ASYNC_TILE_4K(kb + (size_t)nk0 * HD, k_lds[nb2]);
      ASYNC_TILE_4K(vb + (size_t)nk0 * HD, v_lds[nb2]);
      asm volatile("s_wait_asynccnt 0x10" ::: "memory");
    } else {
      asm volatile("s_wait_asynccnt 0x0" ::: "memory");
    }

    const __bf16*   kt_s = &ktile[cur][0];
    const _Float16* vt_s = &vtile[cur][0];

    v8f S[2] = {v8f{}, v8f{}};
#pragma unroll
    for (int f = 0; f < 2; f++) {
      // B frags for S from LDS K tile: lane = column key, contiguous channels
      v16bf B0, B1;
      {
        int c = 16 * f + ln;
        int kc = hl * 16;
        const __bf16* p0 = kt_s + c * 64 + kc;        // channels 0..31 half
        const __bf16* p1 = kt_s + c * 64 + 32 + kc;   // channels 32..63 half
        v8bf a0 = *(const v8bf*)p0, a1 = *(const v8bf*)(p0 + 8);
        v8bf b0 = *(const v8bf*)p1, b1 = *(const v8bf*)(p1 + 8);
#pragma unroll
        for (int i = 0; i < 8; i++) {
          B0[i] = a0[i]; B0[8 + i] = a1[i];
          B1[i] = b0[i]; B1[8 + i] = b1[i];
        }
      }
      S[f] = WMMA_BF(A0, B0, S[f]);
      S[f] = WMMA_BF(A1, B1, S[f]);
    }
    // scale + separable rel-pos bias + masking, row max over 32 keys
    float pv[2][8], tmax[8];
#pragma unroll
    for (int r = 0; r < 8; r++) {
      int m = r + 8 * hl;
      int q = q0 + m;
      float best = NEGBIG;
#pragma unroll
      for (int f = 0; f < 2; f++) {
        int key = k0 + 16 * f + ln;
        float s = S[f][r] * SCALE_F;
        if (key >= NTOK || q >= NTOK) {
          s = NEGBIG;
        } else if (q >= 1 && key >= 1) {
          int kk = key - 1;
          int tk = kk / (HH * WW), rem = kk % (HH * WW), hk = rem / WW, wk = rem % WW;
          s += relq[m][tk] + relq[m][8 + hk] + relq[m][22 + wk];
        }
        pv[f][r] = s;
        best = fmaxf(best, s);
      }
#pragma unroll
      for (int off = 1; off < 16; off <<= 1) best = fmaxf(best, __shfl_xor(best, off, 32));
      tmax[r] = best;
    }
    // online softmax update; stage P into LDS
#pragma unroll
    for (int r = 0; r < 8; r++) {
      int m = r + 8 * hl;
      float mnew = fmaxf(mrow[r], tmax[r]);
      float alpha = __expf(mrow[r] - mnew);   // -1e30 - finite underflows to 0
      float psum = 0.f;
#pragma unroll
      for (int f = 0; f < 2; f++) {
        float p = __expf(pv[f][r] - mnew);
        psum += p;
        pt[m * 32 + 16 * f + ln] = (_Float16)p;
      }
#pragma unroll
      for (int off = 1; off < 16; off <<= 1) psum += __shfl_xor(psum, off, 32);
      lrow[r] = lrow[r] * alpha + psum;
      mrow[r] = mnew;
#pragma unroll
      for (int j = 0; j < 4; j++) O[j][r] *= alpha;
    }
    __syncthreads();
    asm volatile("s_wait_dscnt 0x0" ::: "memory");
    // P (16x32 f16) as A fragment from LDS
    v16h Pa;
    {
      int kh = hl * 16;
#pragma unroll
      for (int i = 0; i < 8; i++) {
        Pa[i]     = pt[ln * 32 + kh + i];
        Pa[8 + i] = pt[ln * 32 + 8 + kh + i];
      }
    }
    // O += P @ V  (V as f16 B fragments from LDS tile, column = channel)
#pragma unroll
    for (int j = 0; j < 4; j++) {
      v16h Bv;
      int c = j * 16 + ln;
#pragma unroll
      for (int e = 0; e < 16; e++) {
        Bv[e] = vt_s[(16 * hl + e) * 64 + c];
      }
      O[j] = WMMA_F16(Pa, Bv, O[j]);
    }
    __syncthreads();
  }

  // epilogue: normalize, residual (+q for tokens>=1), write bf16 for proj GEMM
#pragma unroll
  for (int r = 0; r < 8; r++) {
    int m = r + 8 * hl;
    int q = q0 + m;
    if (q < NTOK) {
      float inv = 1.f / lrow[r];
#pragma unroll
      for (int j = 0; j < 4; j++) {
        int c = j * 16 + ln;
        float val = O[j][r] * inv;
        if (q >= 1) val += qpool[((size_t)bh * NTOK + q) * HD + c];
        outbf[((size_t)b * NTOK + q) * COUT + head * HD + c] = (__bf16)val;
      }
    }
  }
}

// ---------------- output projection GEMM: (6276x768)@(768x768)+b -> d_out fp32 ----
__global__ __launch_bounds__(32) void proj_gemm_kernel(
    const __bf16* __restrict__ ab,   // 6276x768 bf16
    const __bf16* __restrict__ wt,   // 768x768 bf16 (pre-transposed)
    const float*  __restrict__ bias,
    float* __restrict__ out) {
  int idx = blockIdx.x;
  int mt = idx / 12;
  int ng = idx % 12;
  int lane = threadIdx.x;
  int hl = lane >> 4;
  int col0 = ng * 64;
  v8f acc[4] = {v8f{}, v8f{}, v8f{}, v8f{}};
  for (int k0 = 0; k0 < 768; k0 += 32) {
    __builtin_prefetch(ab + (size_t)(mt * 16 + (lane & 15)) * 768 + k0 + 64, 0, 1);
    v16bf a = load_a_bf(ab, 768, mt * 16, k0, MTOT);
#pragma unroll
    for (int j = 0; j < 4; j++) {
      v16bf b = load_b_bf(wt, 768, col0 + j * 16, k0, COUT);
      acc[j] = WMMA_BF(a, b, acc[j]);
    }
  }
#pragma unroll
  for (int j = 0; j < 4; j++) {
    int ncol = col0 + j * 16 + (lane & 15);
    float bi = bias[ncol];
#pragma unroll
    for (int r = 0; r < 8; r++) {
      int m = mt * 16 + r + 8 * hl;
      if (m < MTOT) out[(size_t)m * COUT + ncol] = acc[j][r] + bi;
    }
  }
}

// ---------------- host launch ----------------
extern "C" void kernel_launch(void* const* d_in, const int* in_sizes, int n_in,
                              void* d_out, int out_size, void* d_ws, size_t ws_size,
                              hipStream_t stream) {
  (void)in_sizes; (void)n_in; (void)out_size; (void)ws_size;
  const float* x        = (const float*)d_in[0];
  const float* qkv_w    = (const float*)d_in[1];
  const float* qkv_b    = (const float*)d_in[2];
  const float* proj_w   = (const float*)d_in[3];
  const float* proj_b   = (const float*)d_in[4];
  const float* pool_q_w = (const float*)d_in[5];
  const float* pool_k_w = (const float*)d_in[6];
  const float* pool_v_w = (const float*)d_in[7];
  const float* norm_q_w = (const float*)d_in[8];
  const float* norm_q_b = (const float*)d_in[9];
  const float* norm_k_w = (const float*)d_in[10];
  const float* norm_k_b = (const float*)d_in[11];
  const float* norm_v_w = (const float*)d_in[12];
  const float* norm_v_b = (const float*)d_in[13];
  const float* rel_t    = (const float*)d_in[14];
  const float* rel_h    = (const float*)d_in[15];
  const float* rel_w    = (const float*)d_in[16];
  float* out = (float*)d_out;

  char* ws = (char*)d_ws;
  size_t off = 0;
  auto alloc = [&](size_t bytes) -> void* {
    void* p = ws + off;
    off = (off + bytes + 255) & ~(size_t)255;
    return p;
  };
  const size_t NE = (size_t)MTOT * 768;       // 4,819,968 elements
  __bf16*   xb     = (__bf16*)alloc(NE * 2);
  __bf16*   wqkvt  = (__bf16*)alloc((size_t)QKV3 * 768 * 2);
  __bf16*   wprojt = (__bf16*)alloc((size_t)768 * 768 * 2);
  float*    qraw   = (float*)alloc(NE * 4);
  float*    kraw   = (float*)alloc(NE * 4);
  float*    vraw   = (float*)alloc(NE * 4);
  float*    qpool  = (float*)alloc(NE * 4);
  float*    kpool  = (float*)alloc(NE * 4);
  float*    vpool  = (float*)alloc(NE * 4);
  __bf16*   qbfp   = (__bf16*)alloc(NE * 2);
  __bf16*   kbfp   = (__bf16*)alloc(NE * 2 + 4096);   // +pad: async tile tail over-read
  _Float16* vhp    = (_Float16*)alloc(NE * 2 + 4096); // +pad: async tile tail over-read
  float*    reltb  = (float*)alloc((size_t)BB * HEADS * LSP * 8 * 4);
  float*    relhb  = (float*)alloc((size_t)BB * HEADS * LSP * 14 * 4);
  float*    relwb  = (float*)alloc((size_t)BB * HEADS * LSP * 14 * 4);
  __bf16*   aout   = (__bf16*)alloc(NE * 2);

  // 1. fp32 -> bf16 conversions / weight transposes
  {
    int n = (int)NE;
    cvt_bf16_kernel<<<(n + 255) / 256, 256, 0, stream>>>(x, xb, n);
    int n2 = QKV3 * 768;
    transpose_bf16_kernel<<<(n2 + 255) / 256, 256, 0, stream>>>(qkv_w, wqkvt, 768, QKV3);
    int n3 = 768 * 768;
    transpose_bf16_kernel<<<(n3 + 255) / 256, 256, 0, stream>>>(proj_w, wprojt, 768, COUT);
  }
  // 2. QKV GEMM (WMMA bf16)
  qkv_gemm_kernel<<<MTILES * 36, 32, 0, stream>>>(xb, wqkvt, qkv_b, qraw, kraw, vraw);
  // 3. pool + LN for q, k, v
  {
    int g = BB * HEADS * NTOK;
    pool_ln_kernel<<<g, 64, 0, stream>>>(qraw, pool_q_w, norm_q_w, norm_q_b, qpool, qbfp, (_Float16*)nullptr);
    pool_ln_kernel<<<g, 64, 0, stream>>>(kraw, pool_k_w, norm_k_w, norm_k_b, kpool, kbfp, (_Float16*)nullptr);
    pool_ln_kernel<<<g, 64, 0, stream>>>(vraw, pool_v_w, norm_v_w, norm_v_b, vpool, (__bf16*)nullptr, vhp);
  }
  // 4. separable rel-pos projections
  relbias_kernel<<<BB * HEADS * LSP, 64, 0, stream>>>(qpool, rel_t, rel_h, rel_w, reltb, relhb, relwb);
  // 5. flash attention (WMMA bf16 QK^T, WMMA f16 PV, async-LDS K/V staging) + residual
  attn_kernel<<<BB * HEADS * QTILES, 32, 0, stream>>>(qbfp, kbfp, vhp, reltb, relhb, relwb, qpool, aout);
  // 6. output projection (WMMA bf16) -> fp32 d_out
  proj_gemm_kernel<<<MTILES * 12, 32, 0, stream>>>(aout, wprojt, proj_b, out);
}
